// MultiHeadAttentionChunk_77524159693113
// MI455X (gfx1250) — matrix operand order repaired and checked by reading, not compiled, not run on hardware
//
#include <hip/hip_runtime.h>
#include <hip/hip_bf16.h>
#include <stdint.h>

typedef __attribute__((ext_vector_type(16))) _Float16 half16;
typedef __attribute__((ext_vector_type(8)))  _Float16 half8;
typedef __attribute__((ext_vector_type(8)))  float    float8;

#define NHEAD   8
#define DHEAD   64
#define DMODEL  512
#define SEQ     4096
#define BATCH   8
#define CHUNKSZ 256
#define NCHUNK  16
#define MTOT    (BATCH * SEQ)   // 32768

#if __has_builtin(__builtin_amdgcn_global_load_async_to_lds_b128) && \
    __has_builtin(__builtin_amdgcn_s_wait_asynccnt)
#define HAVE_ASYNC_LDS 1
#endif

static __device__ inline half16 join16(half8 lo, half8 hi) {
  half16 r;
#pragma unroll
  for (int i = 0; i < 8; ++i) { r[i] = lo[i]; r[i + 8] = hi[i]; }
  return r;
}

static __device__ inline float8 wmma_f16(half16 a, half16 b, float8 c) {
  // D = A(16x32) x B(32x16) + C, fp32 accumulate
  return __builtin_amdgcn_wmma_f32_16x16x32_f16(false, a, false, b, (short)0, c,
                                                false, false);
}

#ifdef HAVE_ASYNC_LDS
typedef int v4i_ __attribute__((vector_size(16)));
typedef __attribute__((address_space(1))) v4i_* async_gptr;
typedef __attribute__((address_space(3))) v4i_* async_lptr;

static __device__ inline void async_copy16(const _Float16* g, _Float16* l) {
  async_gptr gp = (async_gptr)(uintptr_t)g;
  async_lptr lp = (async_lptr)(uint32_t)(uintptr_t)l;
  __builtin_amdgcn_global_load_async_to_lds_b128(gp, lp, 0, 0);
}
#endif

// ---------------------------------------------------------------------------
// fp32 -> f16 conversion, 8 elements / thread
// ---------------------------------------------------------------------------
__global__ __launch_bounds__(256) void cvt_f32_f16(const float* __restrict__ in,
                                                   _Float16* __restrict__ out,
                                                   int n8) {
  int i = blockIdx.x * 256 + threadIdx.x;
  if (i >= n8) return;
  const float4* p = (const float4*)in;
  float4 a = p[2 * i], b = p[2 * i + 1];
  half8 h;
  h[0] = (_Float16)a.x; h[1] = (_Float16)a.y; h[2] = (_Float16)a.z; h[3] = (_Float16)a.w;
  h[4] = (_Float16)b.x; h[5] = (_Float16)b.y; h[6] = (_Float16)b.z; h[7] = (_Float16)b.w;
  ((half8*)out)[i] = h;
}

// ---------------------------------------------------------------------------
// Y[m,n] = sum_k A[m,k] * W[n,k] + bias[n]    (NT GEMM, K = N = 512)
// block = 256 thr (8 waves); wave tile = 16(M) x 64(N); block = 128 x 64.
// The W k-slice (64 rows x 64 k) is staged in LDS, double-buffered, via
// async-to-LDS copies when available (all 8 waves share the same W tile).
// ---------------------------------------------------------------------------
__global__ __launch_bounds__(256) void gemm_nt(const _Float16* __restrict__ A,
                                               const _Float16* __restrict__ W,
                                               const float* __restrict__ bias,
                                               _Float16* __restrict__ Yh,
                                               float* __restrict__ Yf) {
  __shared__ __align__(16) _Float16 Wl[2][64][64];   // 16 KB double buffer

  const int tid  = threadIdx.x;
  const int lane = tid & 31;
  const int wv   = tid >> 5;
  const int ln = lane & 15, g = lane >> 4;
  const int mBase = blockIdx.x * 128 + wv * 16;
  const int nBase = blockIdx.y * 64;
  const long arow = (long)(mBase + ln) * DMODEL;

  // stage one 64x64 W slice: 512 x 16B chunks, 2 per thread
  auto stage = [&](int buf, int k0) {
#pragma unroll
    for (int q = 0; q < 2; ++q) {
      int c = tid * 2 + q;
      int row = c >> 3, col = (c & 7) * 8;
      const _Float16* src = W + (long)(nBase + row) * DMODEL + k0 + col;
      _Float16* dst = &Wl[buf][row][col];
#ifdef HAVE_ASYNC_LDS
      async_copy16(src, dst);
#else
      *(half8*)dst = *(const half8*)src;
#endif
    }
  };

  float8 acc[4];
#pragma unroll
  for (int t = 0; t < 4; ++t) acc[t] = (float8)0.0f;

  stage(0, 0);
#pragma unroll 1
  for (int ks = 0; ks < 8; ++ks) {
    const int k0 = ks * 64;
    const int buf = ks & 1;
    if (ks + 1 < 8) stage(buf ^ 1, k0 + 64);    // prefetch next slice
#ifdef HAVE_ASYNC_LDS
    // leave the 2 just-issued copies in flight, drain the previous stage
    if (ks + 1 < 8) __builtin_amdgcn_s_wait_asynccnt(2);
    else            __builtin_amdgcn_s_wait_asynccnt(0);
#endif
    __syncthreads();

#pragma unroll
    for (int kk = 0; kk < 2; ++kk) {            // two 32-wide wmma k-steps
      const long ab = arow + k0 + kk * 32;
      half16 af = join16(*(const half8*)(A + ab + g * 16),
                         *(const half8*)(A + ab + g * 16 + 8));
#pragma unroll
      for (int t = 0; t < 4; ++t) {
        const _Float16* wp = &Wl[buf][t * 16 + ln][kk * 32 + g * 16];
        half16 bf = join16(*(const half8*)wp, *(const half8*)(wp + 8));
        acc[t] = wmma_f16(af, bf, acc[t]);
      }
    }
    __syncthreads();                            // protect buf before rewrite
  }

#pragma unroll
  for (int t = 0; t < 4; ++t) {
    int n = nBase + t * 16 + ln;
    float bv = bias[n];
#pragma unroll
    for (int r = 0; r < 8; ++r) {
      long row = mBase + r + 8 * g;
      float v = acc[t][r] + bv;
      if (Yf) Yf[row * DMODEL + n] = v;
      else    Yh[row * DMODEL + n] = (_Float16)v;
    }
  }
}

// ---------------------------------------------------------------------------
// Chunked causal attention. One WG (16 waves) per (b, h, chunk).
// Wave qt owns query rows qt*16..qt*16+15; streams 32-key blocks with
// online softmax. S = Q K^T via WMMA (K frags direct from global),
// P x V via WMMA (V^T staged in LDS, P through per-wave LDS scratch).
// ---------------------------------------------------------------------------
__global__ __launch_bounds__(512) void attn_chunk(const _Float16* __restrict__ Q,
                                                  const _Float16* __restrict__ Kt,
                                                  const _Float16* __restrict__ V,
                                                  _Float16* __restrict__ O) {
  __shared__ __align__(16) _Float16 VT[DHEAD][CHUNKSZ + 8];  // V^T, padded
  __shared__ __align__(16) _Float16 PS[16][16 * 32];         // per-wave P tile

  const int bid = blockIdx.x;
  const int b  = bid / (NHEAD * NCHUNK);
  const int h  = (bid / NCHUNK) % NHEAD;
  const int ci = bid % NCHUNK;
  const long seqBase = (long)b * SEQ + (long)ci * CHUNKSZ;

  // ---- stage V^T into LDS (coalesced global b128 -> scattered ds_store) ----
  for (int i = threadIdx.x; i < CHUNKSZ * 8; i += 512) {
    int j = i >> 3, dc = i & 7;
    half8 vv = *(const half8*)(V + (seqBase + j) * DMODEL + h * DHEAD + dc * 8);
#pragma unroll
    for (int e = 0; e < 8; ++e) VT[dc * 8 + e][j] = vv[e];
  }
  __syncthreads();

  const int wv   = threadIdx.x >> 5;   // query tile index 0..15
  const int lane = threadIdx.x & 31;
  const int ln = lane & 15, g = lane >> 4;
  const int qt = wv;
  const float scale = 0.022097086912079608f;  // 1/sqrt(256) * 1/sqrt(8)

  // Q fragments for dq=64 (two 32-wide k-steps), loaded once
  const _Float16* qp = Q + (seqBase + qt * 16 + ln) * DMODEL + h * DHEAD;
  half16 aq0 = join16(*(const half8*)(qp + g * 16),      *(const half8*)(qp + g * 16 + 8));
  half16 aq1 = join16(*(const half8*)(qp + 32 + g * 16), *(const half8*)(qp + 40 + g * 16));

  float8 o[4];
#pragma unroll
  for (int t = 0; t < 4; ++t) o[t] = (float8)0.0f;
  float mrun[8], lrun[8];
#pragma unroll
  for (int r = 0; r < 8; ++r) { mrun[r] = -1e30f; lrun[r] = 0.0f; }

  const int kt2max = (qt * 16 + 15) >> 5;   // last 32-key block touching diag
  _Float16* pw = &PS[wv][0];

  for (int kt2 = 0; kt2 <= kt2max; ++kt2) {
    const int jBase = kt2 * 32;

    // ---- S block: 16 queries x 32 keys ----
    float8 s0 = (float8)0.0f, s1 = (float8)0.0f;
    {
      const _Float16* kp0 = Kt + (seqBase + jBase + ln)      * DMODEL + h * DHEAD + g * 16;
      const _Float16* kp1 = Kt + (seqBase + jBase + 16 + ln) * DMODEL + h * DHEAD + g * 16;
      half16 b00 = join16(*(const half8*)kp0,        *(const half8*)(kp0 + 8));
      half16 b01 = join16(*(const half8*)(kp0 + 32), *(const half8*)(kp0 + 40));
      half16 b10 = join16(*(const half8*)kp1,        *(const half8*)(kp1 + 8));
      half16 b11 = join16(*(const half8*)(kp1 + 32), *(const half8*)(kp1 + 40));
      s0 = wmma_f16(aq0, b00, s0); s0 = wmma_f16(aq1, b01, s0);
      s1 = wmma_f16(aq0, b10, s1); s1 = wmma_f16(aq1, b11, s1);
    }

    // ---- online softmax (row m = r + 8g lives in 16-lane half g) ----
    const bool diag = (kt2 == kt2max);
    float p0[8], p1[8];
#pragma unroll
    for (int r = 0; r < 8; ++r) {
      float v0 = s0[r] * scale;
      float v1 = s1[r] * scale;
      if (diag) {
        int irow = qt * 16 + r + 8 * g;
        if (jBase + ln > irow)      v0 = -1e30f;
        if (jBase + 16 + ln > irow) v1 = -1e30f;
      }
      float bm = fmaxf(v0, v1);
#pragma unroll
      for (int off = 8; off >= 1; off >>= 1)
        bm = fmaxf(bm, __shfl_xor(bm, off, 16));
      float mnew = fmaxf(mrun[r], bm);
      float corr = __expf(mrun[r] - mnew);
      mrun[r] = mnew;
      p0[r] = __expf(v0 - mnew);
      p1[r] = __expf(v1 - mnew);
      float rs = p0[r] + p1[r];
#pragma unroll
      for (int off = 8; off >= 1; off >>= 1)
        rs += __shfl_xor(rs, off, 16);
      lrun[r] = lrun[r] * corr + rs;
#pragma unroll
      for (int t = 0; t < 4; ++t) o[t][r] *= corr;
      // scatter P (C-layout) into per-wave LDS tile [16][32]
      pw[(r + 8 * g) * 32 + ln]      = (_Float16)p0[r];
      pw[(r + 8 * g) * 32 + 16 + ln] = (_Float16)p1[r];
    }
    asm volatile("s_wait_dscnt 0" ::: "memory");  // wave-local LDS transpose

    // ---- O += P x V : P as A-fragment from LDS, V^T rows as B-fragments ----
    half16 pf = join16(*(const half8*)(pw + ln * 32 + g * 16),
                       *(const half8*)(pw + ln * 32 + g * 16 + 8));
#pragma unroll
    for (int t = 0; t < 4; ++t) {
      const _Float16* vp = &VT[t * 16 + ln][jBase + g * 16];
      half16 bf = join16(*(const half8*)vp, *(const half8*)(vp + 8));
      o[t] = wmma_f16(pf, bf, o[t]);
    }
  }

  // ---- normalize and store ----
  float inv[8];
#pragma unroll
  for (int r = 0; r < 8; ++r) inv[r] = 1.0f / lrun[r];
#pragma unroll
  for (int t = 0; t < 4; ++t) {
#pragma unroll
    for (int r = 0; r < 8; ++r) {
      long row = seqBase + qt * 16 + r + 8 * g;
      O[row * DMODEL + h * DHEAD + t * 16 + ln] = (_Float16)(o[t][r] * inv[r]);
    }
  }
}

// ---------------------------------------------------------------------------
extern "C" void kernel_launch(void* const* d_in, const int* in_sizes, int n_in,
                              void* d_out, int out_size, void* d_ws, size_t ws_size,
                              hipStream_t stream) {
  (void)in_sizes; (void)n_in; (void)out_size; (void)ws_size;
  const float* query = (const float*)d_in[0];
  const float* key   = (const float*)d_in[1];
  const float* value = (const float*)d_in[2];
  const float* Wq = (const float*)d_in[3];
  const float* bq = (const float*)d_in[4];
  const float* Wk = (const float*)d_in[5];
  const float* bk = (const float*)d_in[6];
  const float* Wv = (const float*)d_in[7];
  const float* bv = (const float*)d_in[8];
  const float* Wo = (const float*)d_in[9];
  const float* bo = (const float*)d_in[10];

  const size_t NE = (size_t)MTOT * DMODEL;     // 16,777,216 elems
  const size_t NW = (size_t)DMODEL * DMODEL;   // 262,144 elems

  _Float16* base = (_Float16*)d_ws;
  _Float16* Xq   = base;
  _Float16* Xk   = Xq + NE;
  _Float16* Xv   = Xk + NE;
  _Float16* Wq16 = Xv + NE;
  _Float16* Wk16 = Wq16 + NW;
  _Float16* Wv16 = Wk16 + NW;
  _Float16* Wo16 = Wv16 + NW;
  _Float16* Q16  = Wo16 + NW;
  _Float16* K16  = Q16 + NE;
  _Float16* V16  = K16 + NE;
  _Float16* A16  = Xq;   // Xq dead after Q projection -> reuse for attn out

  dim3 blk(256);
  int nbE = (int)(NE / 8 / 256);
  int nbW = (int)(NW / 8 / 256);
  cvt_f32_f16<<<nbE, blk, 0, stream>>>(query, Xq, (int)(NE / 8));
  cvt_f32_f16<<<nbE, blk, 0, stream>>>(key,   Xk, (int)(NE / 8));
  cvt_f32_f16<<<nbE, blk, 0, stream>>>(value, Xv, (int)(NE / 8));
  cvt_f32_f16<<<nbW, blk, 0, stream>>>(Wq, Wq16, (int)(NW / 8));
  cvt_f32_f16<<<nbW, blk, 0, stream>>>(Wk, Wk16, (int)(NW / 8));
  cvt_f32_f16<<<nbW, blk, 0, stream>>>(Wv, Wv16, (int)(NW / 8));
  cvt_f32_f16<<<nbW, blk, 0, stream>>>(Wo, Wo16, (int)(NW / 8));

  dim3 ggrid(MTOT / 128, DMODEL / 64);
  gemm_nt<<<ggrid, blk, 0, stream>>>(Xq, Wq16, bq, Q16, nullptr);
  gemm_nt<<<ggrid, blk, 0, stream>>>(Xk, Wk16, bk, K16, nullptr);
  gemm_nt<<<ggrid, blk, 0, stream>>>(Xv, Wv16, bv, V16, nullptr);

  attn_chunk<<<BATCH * NHEAD * NCHUNK, 512, 0, stream>>>(Q16, K16, V16, A16);

  gemm_nt<<<ggrid, blk, 0, stream>>>(A16, Wo16, bo, nullptr, (float*)d_out);
}
